// Mamba2Block_55843164782907
// MI455X (gfx1250) — matrix operand order repaired and checked by reading, not compiled, not run on hardware
//
#include <hip/hip_runtime.h>
#include <hip/hip_bf16.h>

typedef __bf16 bf16;
typedef __attribute__((ext_vector_type(16))) __bf16 v16bf;
typedef __attribute__((ext_vector_type(8)))  __bf16 v8bf;
typedef __attribute__((ext_vector_type(8)))  float  v8f;

// ---------------- helpers ----------------
__device__ __forceinline__ bf16 f2bf(float f) {
  unsigned u = __builtin_bit_cast(unsigned, f);
  unsigned r = u + 0x7FFFu + ((u >> 16) & 1u);     // round-to-nearest-even
  unsigned short s = (unsigned short)(r >> 16);
  return __builtin_bit_cast(bf16, s);
}
__device__ __forceinline__ float bf2f(bf16 h) {
  unsigned short s = __builtin_bit_cast(unsigned short, h);
  unsigned u = ((unsigned)s) << 16;
  return __builtin_bit_cast(float, u);
}
__device__ __forceinline__ v8f zero_v8f() {
  v8f z;
#pragma unroll
  for (int i = 0; i < 8; i++) z[i] = 0.f;
  return z;
}
__device__ __forceinline__ v8f wmma_bf16(v16bf a, v16bf b, v8f c) {
  return __builtin_amdgcn_wmma_f32_16x16x32_bf16(false, a, false, b, (short)0, c,
                                                 false, false);
}
// A fragment (16x32 bf16): lane L holds row M=L%16, K in {k0..k0+7, k0+16..k0+23},
// k0 = 8 for upper half-wave.  Two aligned 16B LDS loads.
__device__ __forceinline__ v16bf ld_fragA(const bf16* row, int kk, int lane) {
  const int ks = kk + ((lane & 16) >> 1);
  v8bf lo = *reinterpret_cast<const v8bf*>(row + ks);
  v8bf hi = *reinterpret_cast<const v8bf*>(row + ks + 16);
  v16bf r;
#pragma unroll
  for (int i = 0; i < 8; i++) { r[i] = lo[i]; r[i + 8] = hi[i]; }
  return r;
}
// B fragment (32x16 bf16): lane L holds col N=L%16, K in {k0..k0+7, k0+16..k0+23},
// k0 = 8 for upper half-wave.  Two aligned 16B LDS loads.
__device__ __forceinline__ v16bf ld_fragB(const bf16* row, int kk, int lane) {
  const int ks = kk + ((lane & 16) >> 1);
  v8bf lo = *reinterpret_cast<const v8bf*>(row + ks);
  v8bf hi = *reinterpret_cast<const v8bf*>(row + ks + 16);
  v16bf r;
#pragma unroll
  for (int i = 0; i < 8; i++) { r[i] = lo[i]; r[i + 8] = hi[i]; }
  return r;
}
// async 16B global -> LDS copy (ASYNCcnt-tracked, per-lane)
__device__ __forceinline__ void async_copy_b128(void* lds_dst, const void* gsrc) {
  unsigned la = (unsigned)(size_t)lds_dst;  // low 32 bits = LDS offset
  asm volatile("global_load_async_to_lds_b128 %0, %1, off"
               :: "v"(la), "v"(gsrc)
               : "memory");
}

// ---------------- conversion kernels ----------------
__global__ void cvt_f32_bf16(const float* __restrict__ in, bf16* __restrict__ out, int n) {
  int i = blockIdx.x * 256 + threadIdx.x;
  if (i < n) out[i] = f2bf(in[i]);
}
// in_proj_w (4256x1024) -> bf16 padded to 4352 rows (pad rows zero)
__global__ void cvt_w1(const float* __restrict__ w, bf16* __restrict__ out) {
  int i = blockIdx.x * 256 + threadIdx.x;  // 4352*1024 total
  int row = i >> 10;
  out[i] = (row < 4256) ? f2bf(w[i]) : f2bf(0.f);
}

// ---------------- tiled WMMA GEMM:  C[M,N] = A[M,K] * B[N,K]^T ----------------
// 128x128 block tile, BK=32, double-buffered async global->LDS staging.
#define BM 128
#define BN 128
#define BK 32
#define GLDT 40  // LDS stride (bf16 elems); 80B, multiple of 16B

template <bool OUT_BF16>
__global__ __launch_bounds__(256) void gemm_nt(const bf16* __restrict__ A,
                                               const bf16* __restrict__ Bw,
                                               void* __restrict__ Cout,
                                               int K, int ldc) {
  __shared__ bf16 As[2][BM * GLDT];
  __shared__ bf16 Bs[2][BN * GLDT];
  const int t = threadIdx.x;
  const int lane = t & 31, wid = t >> 5;
  const int bm = blockIdx.y, bn = blockIdx.x;
  const int waveM = wid >> 2, waveN = wid & 3;  // 2 x 4 wave grid
  v8f acc[4][2];
#pragma unroll
  for (int i = 0; i < 4; i++)
#pragma unroll
    for (int j = 0; j < 2; j++) acc[i][j] = zero_v8f();

  const bf16* aBase = A + (size_t)bm * BM * K;
  const bf16* bBase = Bw + (size_t)bn * BN * K;

  // issue one 128x32 bf16 tile of A and of W into LDS buffer `buf`
  // (each thread: 2 x 16B per matrix  ->  4 async instructions per wave)
  auto issue_tile = [&](int buf, int k0) {
#pragma unroll
    for (int j = 0; j < 2; j++) {
      int q = t + 256 * j;
      int row = q >> 2;
      int cc = (q & 3) << 3;
      async_copy_b128(&As[buf][row * GLDT + cc],
                      aBase + (size_t)row * K + k0 + cc);
      async_copy_b128(&Bs[buf][row * GLDT + cc],
                      bBase + (size_t)row * K + k0 + cc);
    }
  };

  issue_tile(0, 0);
  int cur = 0;
  for (int k0 = 0; k0 < K; k0 += BK) {
    if (k0 + BK < K) {
      issue_tile(cur ^ 1, k0 + BK);                  // prefetch next tile
      asm volatile("s_wait_asynccnt 4" ::: "memory");  // previous tile done
    } else {
      asm volatile("s_wait_asynccnt 0" ::: "memory");
    }
    __syncthreads();
    {
      v16bf fa[4], fb[2];
#pragma unroll
      for (int i = 0; i < 4; i++)
        fa[i] = ld_fragA(&As[cur][(waveM * 64 + i * 16 + (lane & 15)) * GLDT], 0, lane);
#pragma unroll
      for (int j = 0; j < 2; j++)
        fb[j] = ld_fragB(&Bs[cur][(waveN * 32 + j * 16 + (lane & 15)) * GLDT], 0, lane);
#pragma unroll
      for (int i = 0; i < 4; i++)
#pragma unroll
        for (int j = 0; j < 2; j++) acc[i][j] = wmma_bf16(fa[i], fb[j], acc[i][j]);
    }
    __syncthreads();
    cur ^= 1;
  }
  // epilogue: C/D layout -> (row = e + 8*(lane/16), col = lane%16) per tile
#pragma unroll
  for (int i = 0; i < 4; i++)
#pragma unroll
    for (int j = 0; j < 2; j++)
#pragma unroll
      for (int e = 0; e < 8; e++) {
        int row = bm * BM + waveM * 64 + i * 16 + e + ((lane & 16) >> 1);
        int col = bn * BN + waveN * 32 + j * 16 + (lane & 15);
        float v = acc[i][j][e];
        if (OUT_BF16)
          ((bf16*)Cout)[(size_t)row * ldc + col] = f2bf(v);
        else
          ((float*)Cout)[(size_t)row * ldc + col] = v;
      }
}

// ---------------- dt = softplus(zxbcdt[..., -32:] + dt_bias) ----------------
__global__ void dt_kernel(const bf16* __restrict__ zx, const float* __restrict__ dt_bias,
                          float* __restrict__ dtb) {
  int i = blockIdx.x * 256 + threadIdx.x;  // 16384*32
  int r = i >> 5, h = i & 31;
  float v = bf2f(zx[(size_t)r * 4352 + 4224 + h]) + dt_bias[h];
  dtb[i] = (v > 20.f) ? v : logf(1.f + __expf(v));
}

// ---------------- causal depthwise conv (width 4) + SiLU ----------------
__global__ void conv_kernel(const bf16* __restrict__ zx, const float* __restrict__ cw,
                            const float* __restrict__ cb, bf16* __restrict__ xconv) {
  int i = blockIdx.x * 256 + threadIdx.x;  // 16384*2176
  int r = i / 2176;
  int c = i - r * 2176;
  int l = r & 2047;
  float acc = cb[c];
#pragma unroll
  for (int k = 0; k < 4; k++) {
    int ls = l - 3 + k;
    if (ls >= 0)
      acc += bf2f(zx[(size_t)(r - 3 + k) * 4352 + 2048 + c]) * cw[c * 4 + k];
  }
  acc = acc / (1.f + __expf(-acc));  // silu
  xconv[i] = f2bf(acc);
}

// ---------------- SSD chunked scan: one block per (b,h), 32 chunks ----------------
#define SLDT 72
__global__ __launch_bounds__(256) void ssd_kernel(const bf16* __restrict__ xconv,
                                                  const float* __restrict__ dtb,
                                                  const float* __restrict__ A_log,
                                                  const float* __restrict__ Dv,
                                                  bf16* __restrict__ yout) {
  __shared__ float sAdt[64], sAcs[64], sDt[64];
  __shared__ bf16 tC[64 * SLDT];     // C  (l,n)
  __shared__ bf16 tB[64 * SLDT];     // B  (s,n)
  __shared__ bf16 tBdT[64 * SLDT];   // (B*decay)^T  (n,s)
  __shared__ bf16 tXT[64 * SLDT];    // (x*dt)^T  (p,s)
  __shared__ bf16 tM[64 * SLDT];     // masked  C B^T  (l,s)
  __shared__ bf16 stateB[64 * SLDT]; // running state  (p,n)

  const int t = threadIdx.x, lane = t & 31, wid = t >> 5;
  const int bh = blockIdx.x;
  const int b = bh >> 5, h = bh & 31;
  const float Ah = -__expf(A_log[h]);
  const float Dh = Dv[h];
  const int t0 = wid * 2;            // this wave's two 16x16 output tiles
  const int trow = t0 >> 2, tc0 = t0 & 3;

  for (int i = t; i < 64 * SLDT; i += 256) stateB[i] = f2bf(0.f);
  __syncthreads();

  for (int c = 0; c < 32; c++) {
    const size_t rbase = (size_t)b * 2048 + c * 64;
    // ---- per-position A*dt ----
    if (t < 64) {
      float dtv = dtb[(rbase + t) * 32 + h];
      sDt[t] = dtv;
      sAdt[t] = Ah * dtv;
    }
    __syncthreads();
    if (t < 64) {  // inclusive cumsum
      float s = 0.f;
      for (int k = 0; k <= t; k++) s += sAdt[k];
      sAcs[t] = s;
    }
    __syncthreads();
    const float acsL = sAcs[63];
    // ---- stage tiles ----
#pragma unroll
    for (int j = 0; j < 16; j++) {
      int e = t + 256 * j;
      int s = e >> 6, q = e & 63;
      const bf16* rowg = xconv + (rbase + s) * 2176;
      float xs = bf2f(rowg[h * 64 + q]);
      tXT[q * SLDT + s] = f2bf(xs * sDt[s]);
      float bv = bf2f(rowg[2048 + q]);
      tB[s * SLDT + q] = f2bf(bv);
      tBdT[q * SLDT + s] = f2bf(bv * __expf(acsL - sAcs[s]));
      float cv = bf2f(rowg[2112 + q]);
      tC[s * SLDT + q] = f2bf(cv);
    }
    __syncthreads();
    // ---- matmul1: G = C @ B^T, mask with L = exp(Acs[l]-Acs[s]) (s<=l) ----
    {
      v8f a0 = zero_v8f(), a1 = zero_v8f();
#pragma unroll
      for (int kk = 0; kk < 64; kk += 32) {
        v16bf fa = ld_fragA(tC + (trow * 16 + (lane & 15)) * SLDT, kk, lane);
        v16bf fb0 = ld_fragB(tB + (tc0 * 16 + (lane & 15)) * SLDT, kk, lane);
        v16bf fb1 = ld_fragB(tB + ((tc0 + 1) * 16 + (lane & 15)) * SLDT, kk, lane);
        a0 = wmma_bf16(fa, fb0, a0);
        a1 = wmma_bf16(fa, fb1, a1);
      }
#pragma unroll
      for (int e = 0; e < 8; e++) {
        int l = trow * 16 + e + ((lane & 16) >> 1);
        int s0 = tc0 * 16 + (lane & 15);
        int s1 = s0 + 16;
        float f0 = (s0 <= l) ? __expf(sAcs[l] - sAcs[s0]) : 0.f;
        float f1 = (s1 <= l) ? __expf(sAcs[l] - sAcs[s1]) : 0.f;
        tM[l * SLDT + s0] = f2bf(a0[e] * f0);
        tM[l * SLDT + s1] = f2bf(a1[e] * f1);
      }
    }
    __syncthreads();
    // ---- Y = exp(Acs[l]) * (C @ state^T)  +  M @ xdt  +  D * xs ----
    {
      v8f a0 = zero_v8f(), a1 = zero_v8f();
#pragma unroll
      for (int kk = 0; kk < 64; kk += 32) {  // C @ state^T (K = n)
        v16bf fa = ld_fragA(tC + (trow * 16 + (lane & 15)) * SLDT, kk, lane);
        v16bf fb0 = ld_fragB(stateB + (tc0 * 16 + (lane & 15)) * SLDT, kk, lane);
        v16bf fb1 = ld_fragB(stateB + ((tc0 + 1) * 16 + (lane & 15)) * SLDT, kk, lane);
        a0 = wmma_bf16(fa, fb0, a0);
        a1 = wmma_bf16(fa, fb1, a1);
      }
#pragma unroll
      for (int e = 0; e < 8; e++) {  // row scale by exp(Acs[l])
        int l = trow * 16 + e + ((lane & 16) >> 1);
        float sc = __expf(sAcs[l]);
        a0[e] *= sc;
        a1[e] *= sc;
      }
#pragma unroll
      for (int kk = 0; kk < 64; kk += 32) {  // + M @ xdt (K = s)
        v16bf fa = ld_fragA(tM + (trow * 16 + (lane & 15)) * SLDT, kk, lane);
        v16bf fb0 = ld_fragB(tXT + (tc0 * 16 + (lane & 15)) * SLDT, kk, lane);
        v16bf fb1 = ld_fragB(tXT + ((tc0 + 1) * 16 + (lane & 15)) * SLDT, kk, lane);
        a0 = wmma_bf16(fa, fb0, a0);
        a1 = wmma_bf16(fa, fb1, a1);
      }
#pragma unroll
      for (int e = 0; e < 8; e++) {
        int l = trow * 16 + e + ((lane & 16) >> 1);
        int p0 = tc0 * 16 + (lane & 15);
        int p1 = p0 + 16;
        const bf16* rowg = xconv + (rbase + l) * 2176 + h * 64;
        float y0 = a0[e] + Dh * bf2f(rowg[p0]);
        float y1 = a1[e] + Dh * bf2f(rowg[p1]);
        yout[(rbase + l) * 2048 + h * 64 + p0] = f2bf(y0);
        yout[(rbase + l) * 2048 + h * 64 + p1] = f2bf(y1);
      }
    }
    __syncthreads();  // everyone done reading stateB before it is updated
    // ---- state update: state = exp(Acs[63])*state + xdt^T @ (B*decay) ----
    {
      v8f a0 = zero_v8f(), a1 = zero_v8f();
#pragma unroll
      for (int kk = 0; kk < 64; kk += 32) {  // (p,s) @ (n,s)^T
        v16bf fa = ld_fragA(tXT + (trow * 16 + (lane & 15)) * SLDT, kk, lane);
        v16bf fb0 = ld_fragB(tBdT + (tc0 * 16 + (lane & 15)) * SLDT, kk, lane);
        v16bf fb1 = ld_fragB(tBdT + ((tc0 + 1) * 16 + (lane & 15)) * SLDT, kk, lane);
        a0 = wmma_bf16(fa, fb0, a0);
        a1 = wmma_bf16(fa, fb1, a1);
      }
      float dl = __expf(acsL);
#pragma unroll
      for (int e = 0; e < 8; e++) {
        int p = trow * 16 + e + ((lane & 16) >> 1);
        int n0 = tc0 * 16 + (lane & 15);
        int n1 = n0 + 16;
        stateB[p * SLDT + n0] = f2bf(dl * bf2f(stateB[p * SLDT + n0]) + a0[e]);
        stateB[p * SLDT + n1] = f2bf(dl * bf2f(stateB[p * SLDT + n1]) + a1[e]);
      }
    }
    __syncthreads();
  }
}

// ---------------- y = rmsnorm(y * silu(z)) * rms_w   (in-place on y) ----------------
__global__ __launch_bounds__(256) void gate_rms(bf16* __restrict__ y,
                                                const bf16* __restrict__ zx,
                                                const float* __restrict__ rw) {
  __shared__ float red[8];
  int r = blockIdx.x, t = threadIdx.x;
  float v[8];
  float ss = 0.f;
#pragma unroll
  for (int j = 0; j < 8; j++) {
    int c = j * 256 + t;
    float z = bf2f(zx[(size_t)r * 4352 + c]);
    float yv = bf2f(y[(size_t)r * 2048 + c]) * (z / (1.f + __expf(-z)));
    v[j] = yv;
    ss += yv * yv;
  }
#pragma unroll
  for (int m = 16; m > 0; m >>= 1) ss += __shfl_xor(ss, m, 32);
  if ((t & 31) == 0) red[t >> 5] = ss;
  __syncthreads();
  if (t == 0) {
    float s = 0.f;
    for (int k = 0; k < 8; k++) s += red[k];
    red[0] = rsqrtf(s / 2048.f + 1e-5f);
  }
  __syncthreads();
  float rms = red[0];
#pragma unroll
  for (int j = 0; j < 8; j++) {
    int c = j * 256 + t;
    y[(size_t)r * 2048 + c] = f2bf(v[j] * rms * rw[c]);
  }
}

// ---------------- out = layernorm(mm + x) * gamma + beta ----------------
__global__ __launch_bounds__(256) void final_ln(const float* __restrict__ mm,
                                                const float* __restrict__ x,
                                                const float* __restrict__ g,
                                                const float* __restrict__ be,
                                                float* __restrict__ out) {
  __shared__ float redS[8], redQ[8], bc[2];
  int r = blockIdx.x, t = threadIdx.x;
  float hv[4];
  float s = 0.f, sq = 0.f;
#pragma unroll
  for (int j = 0; j < 4; j++) {
    int c = j * 256 + t;
    float v = mm[(size_t)r * 1024 + c] + x[(size_t)r * 1024 + c];
    hv[j] = v;
    s += v;
    sq += v * v;
  }
#pragma unroll
  for (int m = 16; m > 0; m >>= 1) {
    s += __shfl_xor(s, m, 32);
    sq += __shfl_xor(sq, m, 32);
  }
  if ((t & 31) == 0) { redS[t >> 5] = s; redQ[t >> 5] = sq; }
  __syncthreads();
  if (t == 0) {
    float S = 0.f, Q = 0.f;
    for (int k = 0; k < 8; k++) { S += redS[k]; Q += redQ[k]; }
    float mu = S / 1024.f;
    float var = Q / 1024.f - mu * mu;
    bc[0] = mu;
    bc[1] = rsqrtf(var + 1e-5f);
  }
  __syncthreads();
  float mu = bc[0], rstd = bc[1];
#pragma unroll
  for (int j = 0; j < 4; j++) {
    int c = j * 256 + t;
    out[(size_t)r * 1024 + c] = (hv[j] - mu) * rstd * g[c] + be[c];
  }
}

// ---------------- launch ----------------
extern "C" void kernel_launch(void* const* d_in, const int* in_sizes, int n_in,
                              void* d_out, int out_size, void* d_ws, size_t ws_size,
                              hipStream_t stream) {
  const float* x         = (const float*)d_in[0];
  const float* in_proj_w = (const float*)d_in[1];
  const float* conv_w    = (const float*)d_in[2];
  const float* conv_b    = (const float*)d_in[3];
  const float* dt_bias   = (const float*)d_in[4];
  const float* A_log     = (const float*)d_in[5];
  const float* Dv        = (const float*)d_in[6];
  const float* rms_w     = (const float*)d_in[7];
  const float* out_proj_w= (const float*)d_in[8];
  const float* ln_g      = (const float*)d_in[9];
  const float* ln_b      = (const float*)d_in[10];

  char* ws = (char*)d_ws;
  size_t off = 0;
  auto alloc = [&](size_t bytes) -> void* {
    void* p = ws + off;
    off += (bytes + 255) & ~(size_t)255;
    return p;
  };
  bf16*  x_bf  = (bf16*)alloc(16384ull * 1024 * 2);
  bf16*  w1_bf = (bf16*)alloc(4352ull * 1024 * 2);
  bf16*  w2_bf = (bf16*)alloc(1024ull * 2048 * 2);
  bf16*  zx    = (bf16*)alloc(16384ull * 4352 * 2);   // z | xBC | dt   (row stride 4352)
  float* dtb   = (float*)alloc(16384ull * 32 * 4);
  bf16*  xconv = (bf16*)alloc(16384ull * 2176 * 2);   // xs | B | C
  bf16*  ybuf  = (bf16*)alloc(16384ull * 2048 * 2);
  float* mm    = (float*)alloc(16384ull * 1024 * 4);

  cvt_f32_bf16<<<(16384 * 1024) / 256, 256, 0, stream>>>(x, x_bf, 16384 * 1024);
  cvt_w1<<<(4352 * 1024) / 256, 256, 0, stream>>>(in_proj_w, w1_bf);
  cvt_f32_bf16<<<(1024 * 2048) / 256, 256, 0, stream>>>(out_proj_w, w2_bf, 1024 * 2048);

  gemm_nt<true><<<dim3(4352 / BN, 16384 / BM), 256, 0, stream>>>(x_bf, w1_bf, zx,
                                                                 1024, 4352);
  dt_kernel<<<(16384 * 32) / 256, 256, 0, stream>>>(zx, dt_bias, dtb);
  conv_kernel<<<(16384 * 2176) / 256, 256, 0, stream>>>(zx, conv_w, conv_b, xconv);
  ssd_kernel<<<256, 256, 0, stream>>>(xconv, dtb, A_log, Dv, ybuf);
  gate_rms<<<16384, 256, 0, stream>>>(ybuf, zx, rms_w);
  gemm_nt<false><<<dim3(1024 / BN, 16384 / BM), 256, 0, stream>>>(ybuf, w2_bf, mm,
                                                                  2048, 1024);
  final_ln<<<16384, 256, 0, stream>>>(mm, x, ln_g, ln_b, (float*)d_out);
}
